// RESEGNN_36249523978730
// MI455X (gfx1250) — matrix operand order repaired and checked by reading, not compiled, not run on hardware
//
#include <hip/hip_runtime.h>
#include <hip/hip_bf16.h>

#define N_NODESC 10000
#define N_EDGESC 320000
#define HIDC 128
#define K1PAD 288   // 2*HID + 2 (radial, edge_attr) padded 258 -> 288 (multiple of 32)
#define NLAYERS 4
#define EPB 64      // edges per block (4 x 16-row WMMA sub-tiles share each B fragment)

typedef __attribute__((ext_vector_type(16))) _Float16 v16h;
typedef __attribute__((ext_vector_type(8)))  float    v8f;

__device__ __forceinline__ v8f wmma_step(v16h a, v16h b, v8f c) {
  // D = A(16x32 f16) * B(32x16 f16) + C(16x16 f32)
  return __builtin_amdgcn_wmma_f32_16x16x32_f16(false, a, false, b, (short)0, c, false, false);
}

// A fragment: 16x32 f16 tile from row-major [16][ldk] memory (LDS).
// Lane L: row = L&15, K = k0 + 16*(L>>4) .. +15.
__device__ __forceinline__ v16h frag_a(const _Float16* m, int ldk, int k0, int lane) {
  const int row = lane & 15;
  const int kb  = k0 + ((lane >> 4) << 4);
  const _Float16* p = m + row * ldk + kb;
  v16h a;
#pragma unroll
  for (int i = 0; i < 16; ++i) a[i] = p[i];
  return a;
}

// B fragment: 32x16 f16 tile; weights stored transposed [N][K] row-major so the
// 16 consecutive K values per lane are one contiguous 32B load.
// Lane L: col n = n0 + (L&15); K = k0 + 16*(L>>4) .. +15.
__device__ __forceinline__ v16h frag_b(const _Float16* wt, int ldk, int n0, int k0, int lane) {
  const int n  = n0 + (lane & 15);
  const int kb = k0 + ((lane >> 4) << 4);
  const _Float16* p = wt + (size_t)n * ldk + kb;
  v16h b;
#pragma unroll
  for (int i = 0; i < 16; ++i) b[i] = p[i];
  return b;
}

__device__ __forceinline__ float silu_f(float x) { return x / (1.f + __expf(-x)); }

// ---------------------------------------------------------------- weight prep
__global__ void transpose_pad_f16(const float* __restrict__ W, _Float16* __restrict__ Wt,
                                  int K, int N, int Kpad) {
  int idx = blockIdx.x * blockDim.x + threadIdx.x;
  if (idx >= N * Kpad) return;
  int n = idx / Kpad, k = idx - n * Kpad;
  Wt[idx] = (k < K) ? (_Float16)W[(size_t)k * N + n] : (_Float16)0.f;
}

__global__ void zero_f32(float* __restrict__ a, int n) {
  int i = blockIdx.x * blockDim.x + threadIdx.x;
  if (i < n) a[i] = 0.f;
}

// ---------------------------------------------------------------- embedding
__global__ __launch_bounds__(128) void embed_kernel(
    const float* __restrict__ nf, const float* __restrict__ W, const float* __restrict__ b,
    const float* __restrict__ pos_in, float* __restrict__ h, _Float16* __restrict__ hf,
    float* __restrict__ pos) {
  const int i = blockIdx.x, j = threadIdx.x;
  float acc = b[j];
#pragma unroll
  for (int k = 0; k < 18; ++k) acc += nf[i * 18 + k] * W[k * HIDC + j];
  h[(size_t)i * HIDC + j]  = acc;
  hf[(size_t)i * HIDC + j] = (_Float16)acc;
  if (j < 3) pos[i * 3 + j] = pos_in[i * 3 + j];
}

// ---------------------------------------------------------------- edge kernel
// 256 threads = 8 waves; EPB=64 edges per block. Wave w owns output cols
// [16w,16w+16) and sweeps 4 row sub-tiles, sharing one B fragment per k-step.
__global__ __launch_bounds__(256) void egnn_edge_kernel(
    const int* __restrict__ ei, const float* __restrict__ eattr,
    const float* __restrict__ pos, const _Float16* __restrict__ hf,
    const _Float16* __restrict__ We1t, const float* __restrict__ be1,
    const _Float16* __restrict__ We2t, const float* __restrict__ be2,
    const _Float16* __restrict__ Wc1t, const float* __restrict__ bc1,
    const float* __restrict__ Wc2,
    float* __restrict__ agg, float* __restrict__ aggc, float* __restrict__ cnt) {
  __shared__ _Float16 s_in[EPB][K1PAD];
  __shared__ _Float16 s_a1[EPB][HIDC];
  __shared__ _Float16 s_m[EPB][HIDC];
  __shared__ float s_cd[EPB][3];
  __shared__ float s_t[EPB];
  __shared__ int   s_row[EPB];

  const int t = threadIdx.x;
  const int lane = t & 31;
  const int wave = t >> 5;
  const int e0 = blockIdx.x * EPB;

  if (t < EPB) {
    const int e = e0 + t;
    const int r = ei[e];
    const int c = ei[N_EDGESC + e];
    s_row[t] = r;
    const float dx = pos[3 * r + 0] - pos[3 * c + 0];
    const float dy = pos[3 * r + 1] - pos[3 * c + 1];
    const float dz = pos[3 * r + 2] - pos[3 * c + 2];
    s_cd[t][0] = dx; s_cd[t][1] = dy; s_cd[t][2] = dz;
    s_in[t][2 * HIDC + 0] = (_Float16)(dx * dx + dy * dy + dz * dz);
    s_in[t][2 * HIDC + 1] = (_Float16)eattr[e];
#pragma unroll
    for (int k = 2 * HIDC + 2; k < K1PAD; ++k) s_in[t][k] = (_Float16)0.f;
    s_t[t] = 0.f;
  }
  {  // gather h[row], h[col] : thread t -> edge t>>2, 32-half chunk (t&3)*32
    const int e   = t >> 2;
    const int c32 = (t & 3) << 5;
    const int r = ei[e0 + e];
    const int c = ei[N_EDGESC + e0 + e];
    const _Float16* hr = hf + (size_t)r * HIDC + c32;
    const _Float16* hc = hf + (size_t)c * HIDC + c32;
#pragma unroll
    for (int q = 0; q < 32; ++q) { s_in[e][c32 + q] = hr[q]; s_in[e][HIDC + c32 + q] = hc[q]; }
  }
  __syncthreads();

  const int n0 = wave << 4;
  const int nn = n0 + (lane & 15);
  const int mb = (lane >> 4) << 3;

  // GEMM1: [64 x 288] x We1t -> silu -> s_a1
  {
    v8f acc[4];
#pragma unroll
    for (int s = 0; s < 4; ++s) acc[s] = (v8f){0.f, 0.f, 0.f, 0.f, 0.f, 0.f, 0.f, 0.f};
#pragma unroll
    for (int kt = 0; kt < K1PAD / 32; ++kt) {
      const v16h b = frag_b(We1t, K1PAD, n0, kt * 32, lane);
#pragma unroll
      for (int s = 0; s < 4; ++s)
        acc[s] = wmma_step(frag_a(&s_in[16 * s][0], K1PAD, kt * 32, lane), b, acc[s]);
    }
    const float bias = be1[nn];
#pragma unroll
    for (int s = 0; s < 4; ++s)
#pragma unroll
      for (int r = 0; r < 8; ++r) s_a1[16 * s + mb + r][nn] = (_Float16)silu_f(acc[s][r] + bias);
  }
  __syncthreads();

  // GEMM2: x We2t -> silu -> s_m (the edge message)
  {
    v8f acc[4];
#pragma unroll
    for (int s = 0; s < 4; ++s) acc[s] = (v8f){0.f, 0.f, 0.f, 0.f, 0.f, 0.f, 0.f, 0.f};
#pragma unroll
    for (int kt = 0; kt < HIDC / 32; ++kt) {
      const v16h b = frag_b(We2t, HIDC, n0, kt * 32, lane);
#pragma unroll
      for (int s = 0; s < 4; ++s)
        acc[s] = wmma_step(frag_a(&s_a1[16 * s][0], HIDC, kt * 32, lane), b, acc[s]);
    }
    const float bias = be2[nn];
#pragma unroll
    for (int s = 0; s < 4; ++s)
#pragma unroll
      for (int r = 0; r < 8; ++r) s_m[16 * s + mb + r][nn] = (_Float16)silu_f(acc[s][r] + bias);
  }
  __syncthreads();

  // GEMM3: coord-MLP layer 1: m x Wc1t -> silu -> s_a1 (reuse)
  {
    v8f acc[4];
#pragma unroll
    for (int s = 0; s < 4; ++s) acc[s] = (v8f){0.f, 0.f, 0.f, 0.f, 0.f, 0.f, 0.f, 0.f};
#pragma unroll
    for (int kt = 0; kt < HIDC / 32; ++kt) {
      const v16h b = frag_b(Wc1t, HIDC, n0, kt * 32, lane);
#pragma unroll
      for (int s = 0; s < 4; ++s)
        acc[s] = wmma_step(frag_a(&s_m[16 * s][0], HIDC, kt * 32, lane), b, acc[s]);
    }
    const float bias = bc1[nn];
#pragma unroll
    for (int s = 0; s < 4; ++s)
#pragma unroll
      for (int r = 0; r < 8; ++r) s_a1[16 * s + mb + r][nn] = (_Float16)silu_f(acc[s][r] + bias);
  }
  __syncthreads();

  {  // t[e] = dot(s_a1[e][:], Wc2)  (lane-parallel partial sums into LDS)
    const int e  = t >> 2;
    const int j0 = (t & 3) << 5;
    float p = 0.f;
#pragma unroll
    for (int q = 0; q < 32; ++q) p += (float)s_a1[e][j0 + q] * Wc2[j0 + q];
    atomicAdd(&s_t[e], p);
  }
  __syncthreads();

  {  // scatter: agg[row] += m
    const int e  = t >> 2;
    const int j0 = (t & 3) << 5;
    const int r = s_row[e];
    float* ag = agg + (size_t)r * HIDC + j0;
#pragma unroll
    for (int q = 0; q < 32; ++q) atomicAdd(&ag[q], (float)s_m[e][j0 + q]);
  }
  if (t < EPB) {  // scatter: agg_c[row] += coord_diff * t ; cnt[row] += 1
    const int r = s_row[t];
    const float te = s_t[t];
    atomicAdd(&aggc[3 * r + 0], s_cd[t][0] * te);
    atomicAdd(&aggc[3 * r + 1], s_cd[t][1] * te);
    atomicAdd(&aggc[3 * r + 2], s_cd[t][2] * te);
    atomicAdd(&cnt[r], 1.f);
  }
}

// ---------------------------------------------------------------- pos update
__global__ void pos_update_kernel(float* __restrict__ pos, const float* __restrict__ aggc,
                                  const float* __restrict__ cnt, int n3) {
  int i = blockIdx.x * blockDim.x + threadIdx.x;
  if (i < n3) {
    float c = cnt[i / 3];
    pos[i] += aggc[i] / fmaxf(c, 1.f);
  }
}

// ---------------------------------------------------------------- node kernel
// 16 nodes/block, 8 waves: h += silu([h,agg] @ Wn1 + b1) @ Wn2 + b2
__global__ __launch_bounds__(256) void egnn_node_kernel(
    float* __restrict__ h, _Float16* __restrict__ hf, const float* __restrict__ agg,
    const _Float16* __restrict__ Wn1t, const float* __restrict__ bn1,
    const _Float16* __restrict__ Wn2t, const float* __restrict__ bn2) {
  __shared__ _Float16 s_in[16][2 * HIDC];
  __shared__ _Float16 s_a1[16][HIDC];

  const int t = threadIdx.x;
  const int lane = t & 31;
  const int wave = t >> 5;
  const int node0 = blockIdx.x * 16;

  {
    const int i  = t >> 4;
    const int c8 = (t & 15) << 3;
    const int n = node0 + i;
    const _Float16* hr = hf + (size_t)n * HIDC + c8;
    const float*    ar = agg + (size_t)n * HIDC + c8;
#pragma unroll
    for (int q = 0; q < 8; ++q) {
      s_in[i][c8 + q]        = hr[q];
      s_in[i][HIDC + c8 + q] = (_Float16)ar[q];
    }
  }
  __syncthreads();

  const int n0 = wave << 4;

  v8f acc = {0.f, 0.f, 0.f, 0.f, 0.f, 0.f, 0.f, 0.f};
#pragma unroll
  for (int kt = 0; kt < (2 * HIDC) / 32; ++kt)
    acc = wmma_step(frag_a(&s_in[0][0], 2 * HIDC, kt * 32, lane),
                    frag_b(Wn1t, 2 * HIDC, n0, kt * 32, lane), acc);
  {
    const int n = n0 + (lane & 15);
    const float bias = bn1[n];
    const int mb = (lane >> 4) << 3;
#pragma unroll
    for (int r = 0; r < 8; ++r) s_a1[mb + r][n] = (_Float16)silu_f(acc[r] + bias);
  }
  __syncthreads();

  v8f acc2 = {0.f, 0.f, 0.f, 0.f, 0.f, 0.f, 0.f, 0.f};
#pragma unroll
  for (int kt = 0; kt < HIDC / 32; ++kt)
    acc2 = wmma_step(frag_a(&s_a1[0][0], HIDC, kt * 32, lane),
                     frag_b(Wn2t, HIDC, n0, kt * 32, lane), acc2);
  {
    const int n = n0 + (lane & 15);
    const float bias = bn2[n];
    const int mb = (lane >> 4) << 3;
#pragma unroll
    for (int r = 0; r < 8; ++r) {
      const int node = node0 + mb + r;
      float v = h[(size_t)node * HIDC + n] + acc2[r] + bias;
      h[(size_t)node * HIDC + n]  = v;
      hf[(size_t)node * HIDC + n] = (_Float16)v;
    }
  }
}

// ---------------------------------------------------------------- head
__global__ __launch_bounds__(128) void head_kernel(
    const float* __restrict__ h0, const float* __restrict__ Wo, const float* __restrict__ bo,
    const float* __restrict__ Wp1, const float* __restrict__ bp1,
    const float* __restrict__ Wp2, const float* __restrict__ bp2, float* __restrict__ out) {
  __shared__ float central[HIDC];
  __shared__ float relu_v[HIDC];
  const int j = threadIdx.x;
  float a = bo[j];
  for (int k = 0; k < HIDC; ++k) a += h0[k] * Wo[k * HIDC + j];
  central[j] = a;
  __syncthreads();
  float a2 = bp1[j];
  for (int k = 0; k < HIDC; ++k) a2 += central[k] * Wp1[k * HIDC + j];
  relu_v[j] = fmaxf(a2, 0.f);
  __syncthreads();
  if (j < 20) {
    float a3 = bp2[j];
    for (int k = 0; k < HIDC; ++k) a3 += relu_v[k] * Wp2[k * 20 + j];
    out[j] = a3;
  }
}

// ---------------------------------------------------------------- launch
extern "C" void kernel_launch(void* const* d_in, const int* in_sizes, int n_in,
                              void* d_out, int out_size, void* d_ws, size_t ws_size,
                              hipStream_t stream) {
  const float* node_feats = (const float*)d_in[0];
  const float* pos_in     = (const float*)d_in[1];
  const int*   edge_index = (const int*)d_in[2];
  const float* edge_attrs = (const float*)d_in[3];

  // Param leaves in pytree order: Wp1, Wp2, emb_in, emb_out, layers[Wc1,Wc2,We1,We2,Wn1,Wn2]
  const float *Wp1W, *Wp1b, *Wp2W, *Wp2b, *WembW, *Wembb, *WoW, *Wob;
  const float *Wc1W[4], *Wc1b[4], *Wc2v[4], *We1W[4], *We1b[4], *We2W[4], *We2b[4],
              *Wn1W[4], *Wn1b[4], *Wn2W[4], *Wn2b[4];
  if (n_in >= 56) {
    int i = 4;
    Wp1W = (const float*)d_in[i++]; Wp1b = (const float*)d_in[i++];
    Wp2W = (const float*)d_in[i++]; Wp2b = (const float*)d_in[i++];
    WembW = (const float*)d_in[i++]; Wembb = (const float*)d_in[i++];
    WoW = (const float*)d_in[i++]; Wob = (const float*)d_in[i++];
    for (int l = 0; l < NLAYERS; ++l) {
      Wc1W[l] = (const float*)d_in[i++]; Wc1b[l] = (const float*)d_in[i++];
      Wc2v[l] = (const float*)d_in[i++];
      We1W[l] = (const float*)d_in[i++]; We1b[l] = (const float*)d_in[i++];
      We2W[l] = (const float*)d_in[i++]; We2b[l] = (const float*)d_in[i++];
      Wn1W[l] = (const float*)d_in[i++]; Wn1b[l] = (const float*)d_in[i++];
      Wn2W[l] = (const float*)d_in[i++]; Wn2b[l] = (const float*)d_in[i++];
    }
  } else {  // params passed as one flat buffer, same leaf order
    const float* p = (const float*)d_in[4];
    Wp1W = p; p += 128 * 128; Wp1b = p; p += 128;
    Wp2W = p; p += 128 * 20;  Wp2b = p; p += 20;
    WembW = p; p += 18 * 128; Wembb = p; p += 128;
    WoW = p; p += 128 * 128;  Wob = p; p += 128;
    for (int l = 0; l < NLAYERS; ++l) {
      Wc1W[l] = p; p += 128 * 128; Wc1b[l] = p; p += 128;
      Wc2v[l] = p; p += 128;
      We1W[l] = p; p += 258 * 128; We1b[l] = p; p += 128;
      We2W[l] = p; p += 128 * 128; We2b[l] = p; p += 128;
      Wn1W[l] = p; p += 256 * 128; Wn1b[l] = p; p += 128;
      Wn2W[l] = p; p += 128 * 128; Wn2b[l] = p; p += 128;
    }
  }
  (void)in_sizes; (void)out_size; (void)ws_size;

  // Workspace carve-out
  char* ws = (char*)d_ws;
  size_t off = 0;
  auto carve = [&](size_t bytes) -> void* {
    void* r = ws + off;
    off = (off + bytes + 255) & ~(size_t)255;
    return r;
  };
  float*    h    = (float*)carve((size_t)N_NODESC * HIDC * 4);
  _Float16* hf   = (_Float16*)carve((size_t)N_NODESC * HIDC * 2);
  float*    posw = (float*)carve((size_t)N_NODESC * 3 * 4);
  float*    agg  = (float*)carve((size_t)N_NODESC * HIDC * 4);
  float*    aggc = (float*)carve((size_t)N_NODESC * 3 * 4);
  float*    cntb = (float*)carve((size_t)N_NODESC * 4);
  _Float16 *We1t[4], *We2t[4], *Wc1t[4], *Wn1t[4], *Wn2t[4];
  for (int l = 0; l < NLAYERS; ++l) {
    We1t[l] = (_Float16*)carve((size_t)HIDC * K1PAD * 2);
    We2t[l] = (_Float16*)carve((size_t)HIDC * HIDC * 2);
    Wc1t[l] = (_Float16*)carve((size_t)HIDC * HIDC * 2);
    Wn1t[l] = (_Float16*)carve((size_t)HIDC * 2 * HIDC * 2);
    Wn2t[l] = (_Float16*)carve((size_t)HIDC * HIDC * 2);
  }

  const int TB = 256;
  // Weight prep (re-done every call; deterministic)
  for (int l = 0; l < NLAYERS; ++l) {
    transpose_pad_f16<<<(HIDC * K1PAD + TB - 1) / TB, TB, 0, stream>>>(We1W[l], We1t[l], 258, HIDC, K1PAD);
    transpose_pad_f16<<<(HIDC * HIDC + TB - 1) / TB, TB, 0, stream>>>(We2W[l], We2t[l], HIDC, HIDC, HIDC);
    transpose_pad_f16<<<(HIDC * HIDC + TB - 1) / TB, TB, 0, stream>>>(Wc1W[l], Wc1t[l], HIDC, HIDC, HIDC);
    transpose_pad_f16<<<(HIDC * 2 * HIDC + TB - 1) / TB, TB, 0, stream>>>(Wn1W[l], Wn1t[l], 2 * HIDC, HIDC, 2 * HIDC);
    transpose_pad_f16<<<(HIDC * HIDC + TB - 1) / TB, TB, 0, stream>>>(Wn2W[l], Wn2t[l], HIDC, HIDC, HIDC);
  }

  embed_kernel<<<N_NODESC, 128, 0, stream>>>(node_feats, WembW, Wembb, pos_in, h, hf, posw);

  for (int l = 0; l < NLAYERS; ++l) {
    zero_f32<<<(N_NODESC * HIDC + TB - 1) / TB, TB, 0, stream>>>(agg, N_NODESC * HIDC);
    zero_f32<<<(N_NODESC * 3 + TB - 1) / TB, TB, 0, stream>>>(aggc, N_NODESC * 3);
    zero_f32<<<(N_NODESC + TB - 1) / TB, TB, 0, stream>>>(cntb, N_NODESC);

    egnn_edge_kernel<<<N_EDGESC / EPB, 256, 0, stream>>>(
        edge_index, edge_attrs, posw, hf,
        We1t[l], We1b[l], We2t[l], We2b[l], Wc1t[l], Wc1b[l], Wc2v[l],
        agg, aggc, cntb);

    pos_update_kernel<<<(N_NODESC * 3 + TB - 1) / TB, TB, 0, stream>>>(posw, aggc, cntb, N_NODESC * 3);

    egnn_node_kernel<<<N_NODESC / 16, 256, 0, stream>>>(
        h, hf, agg, Wn1t[l], Wn1b[l], Wn2t[l], Wn2b[l]);
  }

  head_kernel<<<1, 128, 0, stream>>>(h, WoW, Wob, Wp1W, Wp1b, Wp2W, Wp2b, (float*)d_out);
}
